// ChainCRF_11647951307300
// MI455X (gfx1250) — matrix operand order repaired and checked by reading, not compiled, not run on hardware
//
#include <hip/hip_runtime.h>

// ---------------------------------------------------------------------------
// ChainCRF loss for MI455X (gfx1250, wave32, WMMA bf16 path, no-LDS GEMM)
// B=128, T=512, D=256, L=21.  N = 441 (trans) + 21 (state), padded to 512.
// S (scores) stored 464 columns wide.
// ---------------------------------------------------------------------------

#define CRF_B    128
#define CRF_T    512
#define CRF_D    256
#define CRF_L    21
#define CRF_NT   441              // L*L transition columns
#define CRF_NS   464              // S leading dimension (29 tiles of 16)
#define CRF_NP   512              // padded GEMM column count (8 blocks of 64)
#define CRF_M    (CRF_B * CRF_T)  // 65536 rows

typedef __attribute__((ext_vector_type(16))) __bf16 v16bf;
typedef __attribute__((ext_vector_type(8)))  float  v8f;

union FragBF { v16bf v; uint4 q[2]; };

__device__ __forceinline__ unsigned int f2bf(float f) {
    unsigned int u = __float_as_uint(f);
    u += 0x7FFFu + ((u >> 16) & 1u);   // round-to-nearest-even bf16
    return u >> 16;
}
__device__ __forceinline__ unsigned int pack2bf(float lo, float hi) {
    return f2bf(lo) | (f2bf(hi) << 16);
}

// ---------------------------------------------------------------------------
// Kernel 1a: convert x (f32, 65536x256) -> xb (bf16), 8 elements / thread.
// ---------------------------------------------------------------------------
__global__ void __launch_bounds__(256)
crf_conv_x(const float* __restrict__ x, unsigned int* __restrict__ xb) {
    const size_t i = (size_t)blockIdx.x * blockDim.x + threadIdx.x;  // 0..2M-1
    const float4* src = (const float4*)x + i * 2;
    const float4 a = src[0];
    const float4 b = src[1];
    uint4 o;
    o.x = pack2bf(a.x, a.y);
    o.y = pack2bf(a.z, a.w);
    o.z = pack2bf(b.x, b.y);
    o.w = pack2bf(b.z, b.w);
    ((uint4*)xb)[i] = o;
}

// ---------------------------------------------------------------------------
// Kernel 1b: build K-major bf16 weights WcT[256][512] + f32 bias[512].
// Column n: n<441 -> trans_W[n], 441<=n<462 -> state_W[n-441], else zero.
// ---------------------------------------------------------------------------
__global__ void crf_prep_weights(const float* __restrict__ state_W,
                                 const float* __restrict__ state_b,
                                 const float* __restrict__ trans_W,
                                 const float* __restrict__ trans_b,
                                 unsigned short* __restrict__ WcT,
                                 float* __restrict__ biasc) {
    const int k = blockIdx.x;                       // 0..255  (D index)
    for (int n = threadIdx.x; n < CRF_NP; n += blockDim.x) {
        float w = 0.f, bb = 0.f;
        if (n < CRF_NT)              { w = trans_W[n * CRF_D + k];             bb = trans_b[n]; }
        else if (n < CRF_NT + CRF_L) { w = state_W[(n - CRF_NT) * CRF_D + k];  bb = state_b[n - CRF_NT]; }
        WcT[(size_t)k * CRF_NP + n] = (unsigned short)f2bf(w);
        if (k == 0) biasc[n] = bb;
    }
}

// ---------------------------------------------------------------------------
// Kernel 2: bf16 WMMA GEMM, fragments loaded directly from global (no LDS).
//   S[m][n] = sum_d xb[m][d] * WcT[d][n] + bias[n]
// Block = 256 threads (8 waves).  Block tile = 256 M x 64 N.
// Wave (wm = wave>>1, wn = wave&1) owns a 64 M x 32 N sub-tile:
//   4 M-tiles x 2 N-tiles = 8 accumulators, 8 v_wmma per K-step of 32.
// A frag (16-bit A 16x32 layout): lane<16 -> M=lane, K chunks [0..7],[16..23];
//                                 lane>=16 -> M=lane-16, K chunks [8..15],[24..31]
//   -> two contiguous 16B loads from the bf16 row.
// B frag (16-bit B 32x16 layout): lane l holds N=l&15, K in the same chunks as A
//   -> strided 16-bit loads from K-major WcT (L1/L2 resident).
// Prefetch of the next K-chunk is unconditional: for the last step it lands at
// the start of the next row (still inside the workspace), which is harmless.
// ---------------------------------------------------------------------------
__global__ void __launch_bounds__(256)
crf_gemm_bf16(const unsigned short* __restrict__ xb,
              const unsigned short* __restrict__ WcT,
              const float* __restrict__ biasc,
              float* __restrict__ S) {
    const int tid  = threadIdx.x;
    const int wave = tid >> 5;
    const int lane = tid & 31;
    const int wm   = wave >> 1;          // 0..3
    const int wn   = wave & 1;           // 0..1

    const int rowW = blockIdx.x * 256 + wm * 64;   // wave's M base
    const int colW = blockIdx.y * 64  + wn * 32;   // wave's N base
    const int lrow = lane & 15;
    const int klo  = (lane < 16) ? 0 : 8;

    v8f acc[4][2];
    #pragma unroll
    for (int m = 0; m < 4; ++m)
        #pragma unroll
        for (int n = 0; n < 2; ++n)
            acc[m][n] = (v8f){};

    #pragma unroll 1
    for (int k0 = 0; k0 < CRF_D; k0 += 32) {
        FragBF A[4], Bf[2];
        #pragma unroll
        for (int m = 0; m < 4; ++m) {
            const unsigned short* pa =
                xb + (size_t)(rowW + m * 16 + lrow) * CRF_D + k0 + klo;
            A[m].q[0] = *(const uint4*)(pa);
            A[m].q[1] = *(const uint4*)(pa + 16);
            __builtin_prefetch(pa + 32, 0, 3);       // global_prefetch_b8
        }
        #pragma unroll
        for (int n = 0; n < 2; ++n) {
            const unsigned short* pb =
                WcT + (size_t)(k0 + klo) * CRF_NP + colW + n * 16 + lrow;
            unsigned int w[8];
            #pragma unroll
            for (int j = 0; j < 8; ++j) {
                const int kk = (j < 4) ? 2 * j : 16 + 2 * (j - 4);
                w[j] = (unsigned int)pb[(size_t)kk * CRF_NP] |
                       ((unsigned int)pb[(size_t)(kk + 1) * CRF_NP] << 16);
            }
            Bf[n].q[0] = make_uint4(w[0], w[1], w[2], w[3]);
            Bf[n].q[1] = make_uint4(w[4], w[5], w[6], w[7]);
        }
        #pragma unroll
        for (int m = 0; m < 4; ++m)
            #pragma unroll
            for (int n = 0; n < 2; ++n)
                acc[m][n] = __builtin_amdgcn_wmma_f32_16x16x32_bf16(
                        false, A[m].v, false, Bf[n].v,
                        (short)0, acc[m][n], false, false);
    }

    // Epilogue: C/D layout: VGPR v -> M = v + 8*(lane>=16), N = lane&15.
    const int mhi = (lane < 16) ? 0 : 8;
    #pragma unroll
    for (int n = 0; n < 2; ++n) {
        const int col = colW + n * 16 + lrow;
        if (col < CRF_NS) {                 // skip zero-padded columns
            const float bv = biasc[col];
            #pragma unroll
            for (int m = 0; m < 4; ++m) {
                const int r0 = rowW + m * 16 + mhi;
                #pragma unroll
                for (int v = 0; v < 8; ++v)
                    S[(size_t)(r0 + v) * CRF_NS + col] = acc[m][n][v] + bv;
            }
        }
    }
}

// ---------------------------------------------------------------------------
// Kernel 3: per-batch gold-path energy + forward (logsumexp) scan + loss.
// One wave32 per batch element; lane j (<21) owns part[j].
// ---------------------------------------------------------------------------
__global__ void __launch_bounds__(32)
crf_scan(const float* __restrict__ S,
         const int* __restrict__ target,
         const float* __restrict__ mask,
         float* __restrict__ loss) {
    const int b = blockIdx.x;
    const int lane = threadIdx.x;
    const float* Sb = S + (size_t)b * CRF_T * CRF_NS;

    // ---- gold path energy: sum_t energy[b,t,prev_t,tgt_t] -------------------
    float te = 0.f;
    for (int t = lane; t < CRF_T; t += 32) {
        const int prev = (t == 0) ? (CRF_L - 1) : target[b * CRF_T + t - 1];
        const int tgt  = target[b * CRF_T + t];
        const float* row = Sb + (size_t)t * CRF_NS;
        te += (row[prev * CRF_L + tgt] + row[CRF_NT + tgt]) * mask[b * CRF_T + t];
    }
    #pragma unroll
    for (int off = 16; off; off >>= 1) te += __shfl_xor(te, off, 32);

    // ---- forward scan -------------------------------------------------------
    float part = -1e30f;
    if (lane < CRF_L) {
        part = (Sb[(CRF_L - 1) * CRF_L + lane] + Sb[CRF_NT + lane]) * mask[b * CRF_T];
        for (int t = 1; t < CRF_T; ++t) {
            const float* row = Sb + (size_t)t * CRF_NS;
            const float mt = mask[b * CRF_T + t];
            const float sj = row[CRF_NT + lane];       // state logit for label j

            float e[CRF_L];
            #pragma unroll
            for (int i = 0; i < CRF_L; ++i)
                e[i] = (row[i * CRF_L + lane] + sj) * mt + __shfl(part, i, 32);

            float mx = e[0];
            #pragma unroll
            for (int i = 1; i < CRF_L; ++i) mx = fmaxf(mx, e[i]);
            float s = 0.f;
            #pragma unroll
            for (int i = 0; i < CRF_L; ++i) s += __expf(e[i] - mx);
            part += (mx + __logf(s) - part) * mt;
        }
    }

    // ---- loss[b] = logsumexp_j(part) - te  (lanes >= 21 hold -1e30) ---------
    float mx = part;
    #pragma unroll
    for (int off = 16; off; off >>= 1) mx = fmaxf(mx, __shfl_xor(mx, off, 32));
    float s = __expf(part - mx);
    #pragma unroll
    for (int off = 16; off; off >>= 1) s += __shfl_xor(s, off, 32);
    if (lane == 0) loss[b] = mx + __logf(s) - te;
}

// ---------------------------------------------------------------------------
extern "C" void kernel_launch(void* const* d_in, const int* in_sizes, int n_in,
                              void* d_out, int out_size, void* d_ws, size_t ws_size,
                              hipStream_t stream) {
    (void)in_sizes; (void)n_in; (void)out_size; (void)ws_size;
    const float* x       = (const float*)d_in[0];
    const float* state_W = (const float*)d_in[1];
    const float* state_b = (const float*)d_in[2];
    const float* trans_W = (const float*)d_in[3];
    const float* trans_b = (const float*)d_in[4];
    const int*   target  = (const int*)  d_in[5];
    const float* mask    = (const float*)d_in[6];
    float* loss = (float*)d_out;

    char* ws = (char*)d_ws;
    unsigned short* WcT   = (unsigned short*)ws;                  // 256*512*2 = 256 KB
    float*          biasc = (float*)(ws + (272 << 10));           // 2 KB
    unsigned short* xb    = (unsigned short*)(ws + (512 << 10));  // 65536*256*2 = 32 MB
    float*          S     = (float*)(ws + (512 << 10) + ((size_t)CRF_M * CRF_D * 2));
                                                                  // 65536*464*4 ≈ 122 MB

    crf_conv_x<<<(CRF_M * CRF_D) / (8 * 256), 256, 0, stream>>>(x, (unsigned int*)xb);

    crf_prep_weights<<<CRF_D, 256, 0, stream>>>(state_W, state_b, trans_W, trans_b,
                                                WcT, biasc);

    dim3 grid(CRF_M / 256, CRF_NP / 64);   // 256 x 8
    crf_gemm_bf16<<<grid, 256, 0, stream>>>(xb, WcT, biasc, S);

    crf_scan<<<CRF_B, 32, 0, stream>>>(S, target, mask, loss);
}
